// RPN_75797582840690
// MI455X (gfx1250) — matrix operand order repaired and checked by reading, not compiled, not run on hardware
//
#include <hip/hip_runtime.h>
#include <hip/hip_bf16.h>
#include <math.h>

typedef __attribute__((ext_vector_type(16))) _Float16 v16h;
typedef __attribute__((ext_vector_type(8)))  _Float16 v8h;
typedef __attribute__((ext_vector_type(8)))  float    v8f;
typedef __attribute__((ext_vector_type(4)))  int      v4i;

#define RH 50
#define RW 38
#define RA 9
#define INC 512
#define MIDC 512
#define KD 4608          // 512*9
#define RM 1900          // 50*38
#define MPAD 1920        // 60 macro-tiles of 32 rows
#define NANCH 17100      // 1900*9
#define NSORT 32768
#define NPRE 12000
#define NPOST 6000
#define IMGH 800.0f
#define IMGW 600.0f
#define MINSZ 16.0f
#define NMS_T 0.7f

#define ASTRIDE 72       // 64 halfs + 8 pad halfs -> 144B row stride, conflict-free ds_load_b128

#if defined(__HIP_DEVICE_COMPILE__) && __has_builtin(__builtin_amdgcn_global_load_async_to_lds_b128) && __has_builtin(__builtin_amdgcn_s_wait_asynccnt)
#define HAS_ASYNC_LDS 1
#else
#define HAS_ASYNC_LDS 0
#endif

// ---------------- weight f32 -> f16 (Bt[o][k], k = c*9 + dy*3 + dx) ----------------
__global__ __launch_bounds__(256) void k_wconv(const float* __restrict__ w, _Float16* __restrict__ bt) {
    int i = blockIdx.x * 256 + threadIdx.x;
    if (i < MIDC * KD) bt[i] = (_Float16)w[i];   // (O,I,3,3) flat == Bt[o][k]
}

// ---------------- im2col to f16 A[MPAD][KD] ----------------
__global__ __launch_bounds__(256) void k_im2col(const float* __restrict__ x, _Float16* __restrict__ A) {
    int e = blockIdx.x * 256 + threadIdx.x;
    if (e >= MPAD * KD) return;
    int p = e / KD, k = e % KD;
    float v = 0.0f;
    if (p < RM) {
        int c = k / 9, r = k % 9;
        int dy = r / 3, dx = r % 3;
        int y = p / RW + dy - 1;
        int xx = p % RW + dx - 1;
        if (y >= 0 && y < RH && xx >= 0 && xx < RW)
            v = x[c * (RH * RW) + y * RW + xx];
    }
    A[e] = (_Float16)v;
}

// ---------------- conv1 as WMMA GEMM ----------------
// Block = 8 waves. Block macro-tile: M=32 (shared A panel, async-staged in LDS), N=256 (32 per wave).
// Wave tile: 32x32 via 2x2 register blocking (4 accumulators, 4 WMMA per 32-K step).
__global__ __launch_bounds__(256) void k_gemm_wmma(const _Float16* __restrict__ A,
                                                   const _Float16* __restrict__ Bt,
                                                   const float* __restrict__ bias,
                                                   float* __restrict__ C) {
    __shared__ _Float16 asub[2][32 * ASTRIDE];

    int tid  = threadIdx.x;
    int wave = tid >> 5;
    int lane = tid & 31;
    int mblk = blockIdx.x >> 1;                 // 60 m macro-tiles
    int nblk = blockIdx.x & 1;                  // 2 n super-tiles of 256
    int m0 = mblk * 32;
    int n0 = nblk * 256 + wave * 32;
    int half = lane >> 4, ln = lane & 15;

    const _Float16* b0row = Bt + (size_t)(n0 + ln) * KD;
    const _Float16* b1row = Bt + (size_t)(n0 + 16 + ln) * KD;

    // async stage: thread t copies 16B: A[m0 + t/8][ (t%8)*8 .. +7 ] of current K-chunk
    int ar = tid >> 3;
    int ac = (tid & 7) * 8;
    const _Float16* asrc = A + (size_t)(m0 + ar) * KD + ac;

    v8f acc00 = {}, acc01 = {}, acc10 = {}, acc11 = {};

#if HAS_ASYNC_LDS
    _Float16* adst0 = &asub[0][ar * ASTRIDE + ac];
    _Float16* adst1 = &asub[1][ar * ASTRIDE + ac];
    // preload chunk 0
    __builtin_amdgcn_global_load_async_to_lds_b128((v4i*)asrc, (v4i*)adst0, 0, 0);
    for (int kk = 0; kk < KD; kk += 64) {
        int p = (kk >> 6) & 1;
        if (kk + 64 < KD) {
            // overlap: DMA next chunk into other buffer while we compute this one
            __builtin_amdgcn_global_load_async_to_lds_b128(
                (v4i*)(asrc + kk + 64), (v4i*)(p ? adst0 : adst1), 0, 0);
            __builtin_amdgcn_s_wait_asynccnt(1);
        } else {
            __builtin_amdgcn_s_wait_asynccnt(0);
        }
        __syncthreads();
        const _Float16* ab = asub[p];
        #pragma unroll
        for (int ki = 0; ki < 64; ki += 32) {
            v8h a0l = *(const v8h*)(ab + ln * ASTRIDE + ki + 8 * half);
            v8h a0h = *(const v8h*)(ab + ln * ASTRIDE + ki + 16 + 8 * half);
            v8h a1l = *(const v8h*)(ab + (16 + ln) * ASTRIDE + ki + 8 * half);
            v8h a1h = *(const v8h*)(ab + (16 + ln) * ASTRIDE + ki + 16 + 8 * half);
            v16h a0 = __builtin_shufflevector(a0l, a0h, 0,1,2,3,4,5,6,7,8,9,10,11,12,13,14,15);
            v16h a1 = __builtin_shufflevector(a1l, a1h, 0,1,2,3,4,5,6,7,8,9,10,11,12,13,14,15);
            v16h b0 = __builtin_shufflevector(*(const v8h*)(b0row + kk + ki + 8 * half), *(const v8h*)(b0row + kk + ki + 16 + 8 * half), 0,1,2,3,4,5,6,7,8,9,10,11,12,13,14,15);
            v16h b1 = __builtin_shufflevector(*(const v8h*)(b1row + kk + ki + 8 * half), *(const v8h*)(b1row + kk + ki + 16 + 8 * half), 0,1,2,3,4,5,6,7,8,9,10,11,12,13,14,15);
            acc00 = __builtin_amdgcn_wmma_f32_16x16x32_f16(false, a0, false, b0, (short)0, acc00, false, false);
            acc01 = __builtin_amdgcn_wmma_f32_16x16x32_f16(false, a0, false, b1, (short)0, acc01, false, false);
            acc10 = __builtin_amdgcn_wmma_f32_16x16x32_f16(false, a1, false, b0, (short)0, acc10, false, false);
            acc11 = __builtin_amdgcn_wmma_f32_16x16x32_f16(false, a1, false, b1, (short)0, acc11, false, false);
        }
        __syncthreads();
    }
#else
    (void)asrc; (void)asub;
    const _Float16* a0row = A + (size_t)(m0 + ln) * KD;
    const _Float16* a1row = A + (size_t)(m0 + 16 + ln) * KD;
    for (int kk = 0; kk < KD; kk += 32) {
        v8h a0l = *(const v8h*)(a0row + kk + 8 * half);
        v8h a0h = *(const v8h*)(a0row + kk + 16 + 8 * half);
        v8h a1l = *(const v8h*)(a1row + kk + 8 * half);
        v8h a1h = *(const v8h*)(a1row + kk + 16 + 8 * half);
        v16h a0 = __builtin_shufflevector(a0l, a0h, 0,1,2,3,4,5,6,7,8,9,10,11,12,13,14,15);
        v16h a1 = __builtin_shufflevector(a1l, a1h, 0,1,2,3,4,5,6,7,8,9,10,11,12,13,14,15);
        v16h b0 = __builtin_shufflevector(*(const v8h*)(b0row + kk + 8 * half), *(const v8h*)(b0row + kk + 16 + 8 * half), 0,1,2,3,4,5,6,7,8,9,10,11,12,13,14,15);
        v16h b1 = __builtin_shufflevector(*(const v8h*)(b1row + kk + 8 * half), *(const v8h*)(b1row + kk + 16 + 8 * half), 0,1,2,3,4,5,6,7,8,9,10,11,12,13,14,15);
        acc00 = __builtin_amdgcn_wmma_f32_16x16x32_f16(false, a0, false, b0, (short)0, acc00, false, false);
        acc01 = __builtin_amdgcn_wmma_f32_16x16x32_f16(false, a0, false, b1, (short)0, acc01, false, false);
        acc10 = __builtin_amdgcn_wmma_f32_16x16x32_f16(false, a1, false, b0, (short)0, acc10, false, false);
        acc11 = __builtin_amdgcn_wmma_f32_16x16x32_f16(false, a1, false, b1, (short)0, acc11, false, false);
    }
#endif

    float bv0 = bias[n0 + ln];
    float bv1 = bias[n0 + 16 + ln];
    #pragma unroll
    for (int r = 0; r < 8; r++) {
        int mA = m0 + 8 * half + r;        // rows of acc0x
        int mB = m0 + 16 + 8 * half + r;   // rows of acc1x
        if (mA < RM) {
            C[(size_t)mA * MIDC + n0 + ln]      = acc00[r] + bv0;
            C[(size_t)mA * MIDC + n0 + 16 + ln] = acc01[r] + bv1;
        }
        if (mB < RM) {
            C[(size_t)mB * MIDC + n0 + ln]      = acc10[r] + bv0;
            C[(size_t)mB * MIDC + n0 + 16 + ln] = acc11[r] + bv1;
        }
    }
}

// ---------------- 1x1 convs (loc 36ch, score 18ch) ----------------
__global__ __launch_bounds__(256) void k_locscore(const float* __restrict__ conv1,
                                                   const float* __restrict__ loc_w, const float* __restrict__ loc_b,
                                                   const float* __restrict__ sc_w,  const float* __restrict__ sc_b,
                                                   float* __restrict__ locbuf, float* __restrict__ scobuf) {
    int t = blockIdx.x * 256 + threadIdx.x;
    if (t >= RM * 54) return;
    int p = t / 54, o = t % 54;
    const float* crow = conv1 + (size_t)p * MIDC;
    const float* wrow;
    float acc;
    float* dst;
    if (o < 36) { wrow = loc_w + o * MIDC; acc = loc_b[o]; dst = locbuf + p * 36 + o; }
    else { int o2 = o - 36; wrow = sc_w + o2 * MIDC; acc = sc_b[o2]; dst = scobuf + p * 18 + o2; }
    for (int c = 0; c < MIDC; c++) acc += crow[c] * wrow[c];
    *dst = acc;
}

// ---------------- decode boxes, clip, min-size, init sort keys ----------------
__global__ __launch_bounds__(256) void k_proposals(const float* __restrict__ locbuf,
                                                    const float* __restrict__ scobuf,
                                                    const float* __restrict__ anchors,
                                                    float* __restrict__ roi,
                                                    float* __restrict__ key, int* __restrict__ val) {
    int i = blockIdx.x * 256 + threadIdx.x;
    if (i >= NSORT) return;
    if (i >= NANCH) { key[i] = -INFINITY; val[i] = 0; return; }
    int p = i / RA, a = i % RA;
    float s0 = scobuf[p * 18 + a * 2 + 0];
    float s1 = scobuf[p * 18 + a * 2 + 1];
    float mx = fmaxf(s0, s1);
    float e0 = expf(s0 - mx), e1 = expf(s1 - mx);
    float fg = e1 / (e0 + e1);
    float dy = locbuf[p * 36 + a * 4 + 0];
    float dx = locbuf[p * 36 + a * 4 + 1];
    float dh = locbuf[p * 36 + a * 4 + 2];
    float dw = locbuf[p * 36 + a * 4 + 3];
    float a0 = anchors[i * 4 + 0], a1 = anchors[i * 4 + 1];
    float a2 = anchors[i * 4 + 2], a3 = anchors[i * 4 + 3];
    float ah = a2 - a0, aw = a3 - a1;
    float acy = a0 + 0.5f * ah, acx = a1 + 0.5f * aw;
    float cy = dy * ah + acy, cx = dx * aw + acx;
    float h = expf(dh) * ah, w = expf(dw) * aw;
    float y1 = fminf(fmaxf(cy - 0.5f * h, 0.0f), IMGH);
    float x1 = fminf(fmaxf(cx - 0.5f * w, 0.0f), IMGW);
    float y2 = fminf(fmaxf(cy + 0.5f * h, 0.0f), IMGH);
    float x2 = fminf(fmaxf(cx + 0.5f * w, 0.0f), IMGW);
    roi[i * 4 + 0] = y1; roi[i * 4 + 1] = x1; roi[i * 4 + 2] = y2; roi[i * 4 + 3] = x2;
    bool valid = (y2 - y1 >= MINSZ) && (x2 - x1 >= MINSZ);
    key[i] = valid ? fg : -INFINITY;
    val[i] = i;
}

// ---------------- bitonic sort pass (descending by key) ----------------
__global__ __launch_bounds__(256) void k_bitonic(float* __restrict__ key, int* __restrict__ val, int k, int j) {
    int i = blockIdx.x * 256 + threadIdx.x;
    int ixj = i ^ j;
    if (i >= NSORT || ixj <= i) return;
    float ki = key[i], kj = key[ixj];
    bool desc = ((i & k) == 0);
    bool sw = desc ? (ki < kj) : (ki > kj);
    if (sw) {
        key[i] = kj; key[ixj] = ki;
        int t = val[i]; val[i] = val[ixj]; val[ixj] = t;
    }
}

// ---------------- gather top NPRE sorted boxes ----------------
__global__ __launch_bounds__(256) void k_gather(const float* __restrict__ key, const int* __restrict__ val,
                                                 const float* __restrict__ roi,
                                                 float* __restrict__ boxs, float* __restrict__ areas,
                                                 int* __restrict__ validf) {
    int i = blockIdx.x * 256 + threadIdx.x;
    if (i >= NPRE) return;
    int idx = val[i];
    float b0 = roi[idx * 4 + 0], b1 = roi[idx * 4 + 1];
    float b2 = roi[idx * 4 + 2], b3 = roi[idx * 4 + 3];
    boxs[i * 4 + 0] = b0; boxs[i * 4 + 1] = b1; boxs[i * 4 + 2] = b2; boxs[i * 4 + 3] = b3;
    areas[i] = (b2 - b0) * (b3 - b1);
    validf[i] = (key[i] > -1e37f) ? 1 : 0;
}

// ---------------- zero output ----------------
__global__ __launch_bounds__(256) void k_zero(float* __restrict__ out, int n) {
    int i = blockIdx.x * 256 + threadIdx.x;
    if (i < n) out[i] = 0.0f;
}

// ---------------- sequential NMS (single workgroup) + compaction ----------------
__global__ __launch_bounds__(1024) void k_nms(const float* __restrict__ boxs, const float* __restrict__ areas,
                                               const int* __restrict__ validf, float* __restrict__ out) {
    __shared__ unsigned char supp[NPRE];
    __shared__ unsigned char keep[NPRE];
    __shared__ float bi[5];
    int tid = threadIdx.x;
    for (int i = tid; i < NPRE; i += 1024) supp[i] = 0;
    __syncthreads();
    for (int i = 0; i < NPRE; i++) {
        if (tid == 0) {
            keep[i] = (unsigned char)(validf[i] && !supp[i]);
            bi[0] = boxs[i * 4 + 0]; bi[1] = boxs[i * 4 + 1];
            bi[2] = boxs[i * 4 + 2]; bi[3] = boxs[i * 4 + 3];
            bi[4] = areas[i];
        }
        __syncthreads();
        if (keep[i]) {
            for (int j = i + 1 + tid; j < NPRE; j += 1024) {
                float yy1 = fmaxf(bi[0], boxs[j * 4 + 0]);
                float xx1 = fmaxf(bi[1], boxs[j * 4 + 1]);
                float yy2 = fminf(bi[2], boxs[j * 4 + 2]);
                float xx2 = fminf(bi[3], boxs[j * 4 + 3]);
                float inter = fmaxf(yy2 - yy1, 0.0f) * fmaxf(xx2 - xx1, 0.0f);
                float iou = inter / (bi[4] + areas[j] - inter + 1e-9f);
                if (iou > NMS_T) supp[j] = 1;
            }
        }
        __syncthreads();
    }
    if (tid == 0) {
        int c = 0;
        for (int i = 0; i < NPRE && c < NPOST; i++) {
            if (keep[i]) {
                out[c * 4 + 0] = boxs[i * 4 + 0];
                out[c * 4 + 1] = boxs[i * 4 + 1];
                out[c * 4 + 2] = boxs[i * 4 + 2];
                out[c * 4 + 3] = boxs[i * 4 + 3];
                c++;
            }
        }
    }
}

extern "C" void kernel_launch(void* const* d_in, const int* in_sizes, int n_in,
                              void* d_out, int out_size, void* d_ws, size_t ws_size,
                              hipStream_t stream) {
    (void)in_sizes; (void)n_in; (void)out_size; (void)ws_size;
    const float* out_map = (const float*)d_in[0];
    const float* conv1_w = (const float*)d_in[1];
    const float* conv1_b = (const float*)d_in[2];
    const float* loc_w   = (const float*)d_in[3];
    const float* loc_b   = (const float*)d_in[4];
    const float* score_w = (const float*)d_in[5];
    const float* score_b = (const float*)d_in[6];
    const float* anchors = (const float*)d_in[7];
    float* out = (float*)d_out;

    char* base = (char*)d_ws;
    size_t off = 0;
    auto carve = [&](size_t bytes) -> char* {
        char* r = base + off;
        off += (bytes + 255) & ~(size_t)255;
        return r;
    };
    _Float16* Aim    = (_Float16*)carve(sizeof(_Float16) * (size_t)MPAD * KD);
    _Float16* Bt     = (_Float16*)carve(sizeof(_Float16) * (size_t)MIDC * KD);
    float*    conv1  = (float*)carve(sizeof(float) * (size_t)RM * MIDC);
    float*    locbuf = (float*)carve(sizeof(float) * (size_t)RM * 36);
    float*    scobuf = (float*)carve(sizeof(float) * (size_t)RM * 18);
    float*    roi    = (float*)carve(sizeof(float) * (size_t)NANCH * 4);
    float*    key    = (float*)carve(sizeof(float) * NSORT);
    int*      val    = (int*)carve(sizeof(int) * NSORT);
    float*    boxs   = (float*)carve(sizeof(float) * (size_t)NPRE * 4);
    float*    areas  = (float*)carve(sizeof(float) * NPRE);
    int*      validf = (int*)carve(sizeof(int) * NPRE);

    k_wconv<<<(MIDC * KD + 255) / 256, 256, 0, stream>>>(conv1_w, Bt);
    k_im2col<<<(MPAD * KD + 255) / 256, 256, 0, stream>>>(out_map, Aim);
    k_gemm_wmma<<<(MPAD / 32) * 2, 256, 0, stream>>>(Aim, Bt, conv1_b, conv1);
    k_locscore<<<(RM * 54 + 255) / 256, 256, 0, stream>>>(conv1, loc_w, loc_b, score_w, score_b, locbuf, scobuf);
    k_proposals<<<NSORT / 256, 256, 0, stream>>>(locbuf, scobuf, anchors, roi, key, val);
    for (int k = 2; k <= NSORT; k <<= 1)
        for (int j = k >> 1; j > 0; j >>= 1)
            k_bitonic<<<NSORT / 256, 256, 0, stream>>>(key, val, k, j);
    k_gather<<<(NPRE + 255) / 256, 256, 0, stream>>>(key, val, roi, boxs, areas, validf);
    k_zero<<<(NPOST * 4 + 255) / 256, 256, 0, stream>>>(out, NPOST * 4);
    k_nms<<<1, 1024, 0, stream>>>(boxs, areas, validf, out);
}
